// LayoutPenalty_90383291777489
// MI455X (gfx1250) — matrix operand order repaired and checked by reading, not compiled, not run on hardware
//
#include <hip/hip_runtime.h>
#include <hip/hip_bf16.h>
#include <stdint.h>

typedef __attribute__((ext_vector_type(16))) _Float16 v16h;
typedef __attribute__((ext_vector_type(8)))  float    v8f;
typedef __attribute__((ext_vector_type(4)))  float    v4f;

#define WAVE 32
#define LOG2E 1.44269504088896340736f

#if __has_builtin(__builtin_amdgcn_exp2f)
#define EXP2F(x) __builtin_amdgcn_exp2f(x)
#else
#define EXP2F(x) exp2f(x)
#endif

#if __has_builtin(__builtin_amdgcn_rcpf)
#define RCPF(x) __builtin_amdgcn_rcpf(x)
#else
#define RCPF(x) (1.0f / (x))
#endif

__global__ void lp_init_out(float* out) {
    if (threadIdx.x < 3) out[threadIdx.x] = 0.0f;
}

// D = A x B + C with A = constant mask matrix (16x32 f16), B = exp data (32x16 f16).
//   A row 0 = digit mask over vocab k, A row 1 = all-ones; rows 2..15 = 0.
//   B column n = exp(logits - max) of tile-row n.
// => D[0][n] = digit mass, D[1][n] = total mass, landing in lane n's c[0]/c[1].
__global__ __launch_bounds__(256)
void LayoutPenalty_kernel(const float* __restrict__ logits,
                          const int*   __restrict__ gt_is_digit,
                          const int*   __restrict__ gt_len,
                          float*       __restrict__ out,
                          int nRows, int kShift, int kMask,
                          int tilesPerWave, int totalTiles, float scale) {
    __shared__ float sP;
    __shared__ int   sV;
    __shared__ int   sC;
    if (threadIdx.x == 0) { sP = 0.0f; sV = 0; sC = 0; }
    __syncthreads();

    const int  lane   = threadIdx.x & (WAVE - 1);
    const int  lanelo = lane & 15;
    const bool hi     = (lane >= 16);
    // Wave-uniform tile index chain -> SGPRs (scalar loads for gt_len).
    const int  waveId = __builtin_amdgcn_readfirstlane(
                            (blockIdx.x * blockDim.x + threadIdx.x) / WAVE);

    // ---- Constant A matrices (16x32 f16 masks), built once per lane ----
    // A layout: lane L<16: M=L, halves0-7 = K0..7,  halves8-15 = K16..23
    //           lane L+16: M=L, halves0-7 = K8..15, halves8-15 = K24..31
    v16h a0 = {};
    v16h a1 = {};
    if (lane == 0) {            // row 0 (digit mask): k=0..7
        #pragma unroll
        for (int h = 0; h < 8; ++h) a0[h] = (_Float16)1.0f;
    } else if (lane == 16) {    // row 0: digits k=8,9
        a0[0] = (_Float16)1.0f; a0[1] = (_Float16)1.0f;
    } else if (lane == 1) {     // row 1 (ones) over k=0..7,16..23 ; vocab 32..35
        #pragma unroll
        for (int h = 0; h < 16; ++h) a0[h] = (_Float16)1.0f;
        #pragma unroll
        for (int h = 0; h < 4;  ++h) a1[h] = (_Float16)1.0f;
    } else if (lane == 17) {    // row 1 (ones) over k=8..15,24..31
        #pragma unroll
        for (int h = 0; h < 16; ++h) a0[h] = (_Float16)1.0f;
    }

    float accP = 0.0f;
    int   accV = 0;
    int   accC = 0;

    const int tile0 = waveId * tilesPerWave;
    for (int ti = 0; ti < tilesPerWave; ++ti) {
        const int tile = tile0 + ti;                  // SGPR
        if (tile >= totalTiles) break;
        const int rowBase = tile * 16;                // SGPR
        if (rowBase >= nRows) break;
        const int r = rowBase + lanelo;

        const float* rowp = logits + (size_t)r * 36;  // 144B rows, 16B aligned
        // B layout: lane<16 holds k=0..7,16..23 of row r, lane>=16 holds k=8..15,24..31.
        const int off = hi ? 8 : 0;
        const v4f f0 = __builtin_nontemporal_load((const v4f*)(rowp + off));
        const v4f f1 = __builtin_nontemporal_load((const v4f*)(rowp + off + 4));
        const v4f f2 = __builtin_nontemporal_load((const v4f*)(rowp + off + 16));
        const v4f f3 = __builtin_nontemporal_load((const v4f*)(rowp + off + 20));
        // k=32..35: same address for the lane pair; load on all lanes (merges).
        const v4f f4 = __builtin_nontemporal_load((const v4f*)(rowp + 32));
        // Prefetch epilogue data early (hi lanes duplicate partner's address).
        const int isd = gt_is_digit[r];

        // Row max: local 20 values, then combine with partner lane (L^16).
        float m = fmaxf(fmaxf(fmaxf(f0.x, f0.y), fmaxf(f0.z, f0.w)),
                        fmaxf(fmaxf(f1.x, f1.y), fmaxf(f1.z, f1.w)));
        m = fmaxf(m, fmaxf(fmaxf(fmaxf(f2.x, f2.y), fmaxf(f2.z, f2.w)),
                           fmaxf(fmaxf(f3.x, f3.y), fmaxf(f3.z, f3.w))));
        m = fmaxf(m, fmaxf(fmaxf(f4.x, f4.y), fmaxf(f4.z, f4.w)));
        m = fmaxf(m, __shfl_xor(m, 16, WAVE));

        // exp(x - m) = exp2(x*log2e - m*log2e): one FMA + one v_exp_f32 each.
        const float nm = -m * LOG2E;
        v16h b0;
        b0[0]  = (_Float16)EXP2F(fmaf(f0.x, LOG2E, nm));
        b0[1]  = (_Float16)EXP2F(fmaf(f0.y, LOG2E, nm));
        b0[2]  = (_Float16)EXP2F(fmaf(f0.z, LOG2E, nm));
        b0[3]  = (_Float16)EXP2F(fmaf(f0.w, LOG2E, nm));
        b0[4]  = (_Float16)EXP2F(fmaf(f1.x, LOG2E, nm));
        b0[5]  = (_Float16)EXP2F(fmaf(f1.y, LOG2E, nm));
        b0[6]  = (_Float16)EXP2F(fmaf(f1.z, LOG2E, nm));
        b0[7]  = (_Float16)EXP2F(fmaf(f1.w, LOG2E, nm));
        b0[8]  = (_Float16)EXP2F(fmaf(f2.x, LOG2E, nm));
        b0[9]  = (_Float16)EXP2F(fmaf(f2.y, LOG2E, nm));
        b0[10] = (_Float16)EXP2F(fmaf(f2.z, LOG2E, nm));
        b0[11] = (_Float16)EXP2F(fmaf(f2.w, LOG2E, nm));
        b0[12] = (_Float16)EXP2F(fmaf(f3.x, LOG2E, nm));
        b0[13] = (_Float16)EXP2F(fmaf(f3.y, LOG2E, nm));
        b0[14] = (_Float16)EXP2F(fmaf(f3.z, LOG2E, nm));
        b0[15] = (_Float16)EXP2F(fmaf(f3.w, LOG2E, nm));
        // b1 halves 0..3 are the only entries multiplied by nonzero A1 mask
        // coefficients (vocab 32..35 on lo lanes). Hi-lane values are finite
        // (exp(x-m) <= 1) and hit only zero mask entries -> no select needed.
        v16h b1 = {};
        b1[0] = (_Float16)EXP2F(fmaf(f4.x, LOG2E, nm));
        b1[1] = (_Float16)EXP2F(fmaf(f4.y, LOG2E, nm));
        b1[2] = (_Float16)EXP2F(fmaf(f4.z, LOG2E, nm));
        b1[3] = (_Float16)EXP2F(fmaf(f4.w, LOG2E, nm));

        v8f c = {};
        c = __builtin_amdgcn_wmma_f32_16x16x32_f16(false, a0, false, b0,
                                                   (short)0, c, false, false);
        c = __builtin_amdgcn_wmma_f32_16x16x32_f16(false, a1, false, b1,
                                                   (short)0, c, false, false);

        // Branch-free epilogue: lanes >=16 carry c==0 -> NaN, masked by `act`.
        const int len = gt_len[rowBase >> kShift];          // SGPR s_load
        const bool act = (!hi) && (((rowBase & kMask) + lanelo) < len);
        const float digit_mass = c[0] * RCPF(c[1]);
        const float wrong = (isd == 1) ? (1.0f - digit_mass) : digit_mass;
        accP += act ? wrong : 0.0f;
        accV += (act && (wrong > 0.5f)) ? 1 : 0;
        accC += act ? 1 : 0;
    }

    // Wave reduction
    #pragma unroll
    for (int off = 16; off >= 1; off >>= 1) {
        accP += __shfl_xor(accP, off, WAVE);
        accV += __shfl_xor(accV, off, WAVE);
        accC += __shfl_xor(accC, off, WAVE);
    }
    if (lane == 0) {
        atomicAdd(&sP, accP);
        atomicAdd(&sV, accV);
        atomicAdd(&sC, accC);
    }
    __syncthreads();
    if (threadIdx.x == 0) {
        atomicAdd(&out[0], sP * scale);
        atomicAdd(&out[1], (float)sV);
        atomicAdd(&out[2], (float)sC);
    }
}

extern "C" void kernel_launch(void* const* d_in, const int* in_sizes, int n_in,
                              void* d_out, int out_size, void* d_ws, size_t ws_size,
                              hipStream_t stream) {
    const float* logits      = (const float*)d_in[0];
    const int*   gt_is_digit = (const int*)d_in[1];
    const int*   gt_len      = (const int*)d_in[2];
    float*       out         = (float*)d_out;

    const int B     = in_sizes[2];          // 32768
    const int K     = in_sizes[1] / B;      // 32 (power of 2, multiple of 16)
    const int nRows = in_sizes[1];          // B*K
    const int totalTiles = (nRows + 15) / 16;

    int kShift = 0;
    while ((1 << kShift) < K) ++kShift;     // K=32 -> 5
    const int kMask = K - 1;

    const int BLOCK = 256;                  // 8 waves
    const int GRID  = 1024;
    const int waves = GRID * (BLOCK / WAVE);
    const int tilesPerWave = (totalTiles + waves - 1) / waves;

    const float scale = 1.0f / (float)B;    // BETA / bsz

    lp_init_out<<<1, 32, 0, stream>>>(out);
    LayoutPenalty_kernel<<<GRID, BLOCK, 0, stream>>>(
        logits, gt_is_digit, gt_len, out,
        nRows, kShift, kMask, tilesPerWave, totalTiles, scale);
}
